// BlockSparseAttention_3685081940647
// MI455X (gfx1250) — matrix operand order repaired and checked by reading, not compiled, not run on hardware
//
#include <hip/hip_runtime.h>

// ---------- types ----------
typedef __attribute__((ext_vector_type(16))) __bf16          v16bf;
typedef __attribute__((ext_vector_type(8)))  __bf16          v8bf;
typedef __attribute__((ext_vector_type(8)))  float           v8f;
typedef __attribute__((ext_vector_type(4)))  unsigned int    v4u;

union BF16x16 {
    v16bf b;
    v4u   q[2];
};

// native fp32 -> bf16 (v_cvt_pk_bf16_f32 path)
__device__ __forceinline__ unsigned short bfbits(float f) {
    __bf16 h = (__bf16)f;
    return __builtin_bit_cast(unsigned short, h);
}

// 16 bf16 lanes from two groups of 8 contiguous bf16 (ushort) values (2x b128)
__device__ __forceinline__ v16bf frag_from_bf(const unsigned short* p0, const unsigned short* p1) {
    BF16x16 t;
    t.q[0] = *(const v4u*)p0;
    t.q[1] = *(const v4u*)p1;
    return t.b;
}

__device__ __forceinline__ v8f wmma_bf16(v16bf a, v16bf b, v8f c) {
    return __builtin_amdgcn_wmma_f32_16x16x32_bf16(false, a, false, b, (short)0, c, false, false);
}

// =====================================================================
// Kernel 0: elementwise fp32 -> bf16 (8 elements / thread, b128 traffic)
// =====================================================================
__global__ __launch_bounds__(256) void cvt_bf16_kernel(
    const float* __restrict__ in, unsigned short* __restrict__ out)
{
    const size_t i = ((size_t)blockIdx.x * 256 + threadIdx.x) * 8;
    v8f f = *(const v8f*)(in + i);
    v8bf o;
#pragma unroll
    for (int j = 0; j < 8; ++j) o[j] = (__bf16)f[j];
    *(v8bf*)(out + i) = o;
}

// =====================================================================
// Kernel 1: QKV projection, all-bf16 operands.  Y = X * W^T + bias.
//   Xbf: [16384, 1024] bf16, Wbf: [1024, 1024] bf16 (N x K)
//   transposed==0 -> Out[((b*16+h)*4096 + s)*64 + d]   (Q and K)
//   transposed==1 -> Out[((b*16+h)*64 + d)*4096 + s]   (V transposed)
// 256 thr (8 waves), tile 128x128, wave tile 64x32.
// =====================================================================
__global__ __launch_bounds__(256) void proj_qkv_kernel(
    const unsigned short* __restrict__ Xbf, const unsigned short* __restrict__ Wbf,
    const float* __restrict__ bias, unsigned short* __restrict__ Out,
    float scale, int transposed)
{
    const int lane = threadIdx.x & 31;
    const int wave = threadIdx.x >> 5;
    const int col  = lane & 15;
    const int half = lane >> 4;
    const int m0 = blockIdx.x * 128 + (wave >> 2) * 64;
    const int n0 = blockIdx.y * 128 + (wave & 3) * 32;

    v8f acc[4][2];
#pragma unroll
    for (int mi = 0; mi < 4; ++mi)
#pragma unroll
        for (int ni = 0; ni < 2; ++ni) acc[mi][ni] = (v8f){};

    for (int kb = 0; kb < 1024; kb += 32) {
        v16bf a[4];
#pragma unroll
        for (int mi = 0; mi < 4; ++mi) {
            const unsigned short* p = Xbf + (size_t)(m0 + 16 * mi + col) * 1024 + kb + 8 * half;
            a[mi] = frag_from_bf(p, p + 16);
        }
#pragma unroll
        for (int ni = 0; ni < 2; ++ni) {
            const unsigned short* p = Wbf + (size_t)(n0 + 16 * ni + col) * 1024 + kb + 8 * half;
            v16bf bfr = frag_from_bf(p, p + 16);
#pragma unroll
            for (int mi = 0; mi < 4; ++mi)
                acc[mi][ni] = wmma_bf16(a[mi], bfr, acc[mi][ni]);
        }
    }

#pragma unroll
    for (int mi = 0; mi < 4; ++mi)
#pragma unroll
        for (int ni = 0; ni < 2; ++ni) {
            const int n  = n0 + 16 * ni + col;
            const float bv = bias[n];
            const int h = n >> 6, d = n & 63;
#pragma unroll
            for (int r = 0; r < 8; ++r) {
                const int m  = m0 + 16 * mi + r + 8 * half;
                const int bi = m >> 12, s = m & 4095;
                const float v = (acc[mi][ni][r] + bv) * scale;
                const size_t o = transposed
                    ? ((size_t)(bi * 16 + h) * 64 + d) * 4096 + s
                    : ((size_t)(bi * 16 + h) * 4096 + s) * 64 + d;
                Out[o] = bfbits(v);
            }
        }
}

// =====================================================================
// Kernel 2: block-sparse attention. One workgroup per (block, h, b).
// 4 waves x 32 query rows. Two flash passes (local, window), summed.
//   Qb/Kb: bf16 [B,H,S,D]  (Q pre-scaled by 0.125)
//   Vt   : bf16 [B,H,D,S]
//   Attn : bf16 [B,S,H,D]  output
// =====================================================================
__global__ __launch_bounds__(128) void attn_kernel(
    const unsigned short* __restrict__ Qb,
    const unsigned short* __restrict__ Kb,
    const unsigned short* __restrict__ Vt,
    unsigned short* __restrict__ Attn)
{
    __shared__ __align__(16) unsigned short ldsP[4][32][128];

    const int lane = threadIdx.x & 31;
    const int wave = threadIdx.x >> 5;
    const int col  = lane & 15;
    const int half = lane >> 4;
    const int blk = blockIdx.x;   // 0..31
    const int h   = blockIdx.y;   // 0..15
    const int b   = blockIdx.z;   // 0..3

    const unsigned short* qp = Qb + (size_t)(b * 16 + h) * 4096 * 64;
    const unsigned short* kp = Kb + (size_t)(b * 16 + h) * 4096 * 64;
    const unsigned short* vp = Vt + (size_t)(b * 16 + h) * 64 * 4096;
    const int q0 = blk * 128 + wave * 32;

    // Q fragments (A-layout): 2 row groups x 2 K(=d) chunks
    v16bf qa[2][2];
#pragma unroll
    for (int mi = 0; mi < 2; ++mi)
#pragma unroll
        for (int ks = 0; ks < 2; ++ks) {
            const unsigned short* p = qp + (size_t)(q0 + 16 * mi + col) * 64 + ks * 32 + 8 * half;
            qa[mi][ks] = frag_from_bf(p, p + 16);
        }

    v8f fin[2][4] = {};

    for (int pass = 0; pass < 2; ++pass) {
        const int lo = pass ? (blk > 0 ? blk - 1 : 0)   : blk;
        const int hi = pass ? (blk < 31 ? blk + 1 : 31) : blk;

        v8f mrow[2], lrow[2], o[2][4];
#pragma unroll
        for (int mi = 0; mi < 2; ++mi) {
#pragma unroll
            for (int r = 0; r < 8; ++r) { mrow[mi][r] = -1e30f; lrow[mi][r] = 0.f; }
#pragma unroll
            for (int ni = 0; ni < 4; ++ni) o[mi][ni] = (v8f){};
        }

        for (int kb2 = lo; kb2 <= hi; ++kb2) {
            const int key0 = kb2 * 128;

            // prefetch next key block (K rows and V columns) -> global_prefetch_b8
            if (kb2 < hi) {
                const int nkey0 = (kb2 + 1) * 128;
                __builtin_prefetch(kp + (size_t)(nkey0 + lane * 4) * 64, 0, 1);
                __builtin_prefetch(vp + (size_t)(lane * 2) * 4096 + nkey0, 0, 1);
            }

#pragma unroll
            for (int mi = 0; mi < 2; ++mi) {
                // scores S = Q K^T for 16 rows x 128 keys
                v8f sf[8];
#pragma unroll
                for (int nj = 0; nj < 8; ++nj) {
                    v8f s = (v8f){};
#pragma unroll
                    for (int ks = 0; ks < 2; ++ks) {
                        const unsigned short* p =
                            kp + (size_t)(key0 + 16 * nj + col) * 64 + ks * 32 + 8 * half;
                        v16bf kf = frag_from_bf(p, p + 16);
                        s = wmma_bf16(qa[mi][ks], kf, s);
                    }
                    sf[nj] = s;
                }
                // running row max
                v8f nm = mrow[mi];
#pragma unroll
                for (int nj = 0; nj < 8; ++nj)
#pragma unroll
                    for (int r = 0; r < 8; ++r) nm[r] = fmaxf(nm[r], sf[nj][r]);
#pragma unroll
                for (int r = 0; r < 8; ++r) {
                    float v = nm[r];
#pragma unroll
                    for (int off = 1; off < 16; off <<= 1)
                        v = fmaxf(v, __shfl_xor(v, off, 16));
                    nm[r] = v;
                }
                // rescale accumulators
                v8f fac;
#pragma unroll
                for (int r = 0; r < 8; ++r) fac[r] = __expf(mrow[mi][r] - nm[r]);
#pragma unroll
                for (int ni = 0; ni < 4; ++ni)
#pragma unroll
                    for (int r = 0; r < 8; ++r) o[mi][ni][r] *= fac[r];
                // P = exp(S - max), row sums, stage P (bf16) to LDS
                v8f rs = (v8f){};
#pragma unroll
                for (int nj = 0; nj < 8; ++nj)
#pragma unroll
                    for (int r = 0; r < 8; ++r) {
                        const float pv = __expf(sf[nj][r] - nm[r]);
                        rs[r] += pv;
                        ldsP[wave][16 * mi + r + 8 * half][16 * nj + col] = bfbits(pv);
                    }
#pragma unroll
                for (int r = 0; r < 8; ++r) {
                    float v = rs[r];
#pragma unroll
                    for (int off = 1; off < 16; off <<= 1) v += __shfl_xor(v, off, 16);
                    lrow[mi][r] = lrow[mi][r] * fac[r] + v;
                }
                mrow[mi] = nm;
            }

            // P * V  (A from LDS, B from transposed V, contiguous)
#pragma unroll
            for (int ks = 0; ks < 4; ++ks) {
                v16bf pa[2];
#pragma unroll
                for (int mi = 0; mi < 2; ++mi) {
                    const unsigned short* pp = &ldsP[wave][16 * mi + col][ks * 32 + 8 * half];
                    pa[mi] = frag_from_bf(pp, pp + 16);
                }
#pragma unroll
                for (int ni = 0; ni < 4; ++ni) {
                    const unsigned short* pv =
                        vp + (size_t)(16 * ni + col) * 4096 + key0 + ks * 32 + 8 * half;
                    v16bf vf = frag_from_bf(pv, pv + 16);
#pragma unroll
                    for (int mi = 0; mi < 2; ++mi)
                        o[mi][ni] = wmma_bf16(pa[mi], vf, o[mi][ni]);
                }
            }
        }

        // finalize this pass: fin += o / l
#pragma unroll
        for (int mi = 0; mi < 2; ++mi)
#pragma unroll
            for (int ni = 0; ni < 4; ++ni)
#pragma unroll
                for (int r = 0; r < 8; ++r)
                    fin[mi][ni][r] += o[mi][ni][r] / lrow[mi][r];
    }

    // store attn output bf16 [B,S,H,D]
#pragma unroll
    for (int mi = 0; mi < 2; ++mi)
#pragma unroll
        for (int ni = 0; ni < 4; ++ni)
#pragma unroll
            for (int r = 0; r < 8; ++r) {
                const int s = q0 + 16 * mi + r + 8 * half;
                const int d = 16 * ni + col;
                Attn[(((size_t)b * 4096 + s) * 16 + h) * 64 + d] = bfbits(fin[mi][ni][r]);
            }
}

// =====================================================================
// Kernel 3: output projection. Y = Attn(bf16) * Wo^T(bf16) + bo -> fp32.
// Same tiling as proj_qkv.
// =====================================================================
__global__ __launch_bounds__(256) void proj_out_kernel(
    const unsigned short* __restrict__ Xbf, const unsigned short* __restrict__ Wbf,
    const float* __restrict__ bias, float* __restrict__ Out)
{
    const int lane = threadIdx.x & 31;
    const int wave = threadIdx.x >> 5;
    const int col  = lane & 15;
    const int half = lane >> 4;
    const int m0 = blockIdx.x * 128 + (wave >> 2) * 64;
    const int n0 = blockIdx.y * 128 + (wave & 3) * 32;

    v8f acc[4][2];
#pragma unroll
    for (int mi = 0; mi < 4; ++mi)
#pragma unroll
        for (int ni = 0; ni < 2; ++ni) acc[mi][ni] = (v8f){};

    for (int kb = 0; kb < 1024; kb += 32) {
        v16bf a[4];
#pragma unroll
        for (int mi = 0; mi < 4; ++mi) {
            const unsigned short* p = Xbf + (size_t)(m0 + 16 * mi + col) * 1024 + kb + 8 * half;
            a[mi] = frag_from_bf(p, p + 16);
        }
#pragma unroll
        for (int ni = 0; ni < 2; ++ni) {
            const unsigned short* p = Wbf + (size_t)(n0 + 16 * ni + col) * 1024 + kb + 8 * half;
            v16bf bfr = frag_from_bf(p, p + 16);
#pragma unroll
            for (int mi = 0; mi < 4; ++mi)
                acc[mi][ni] = wmma_bf16(a[mi], bfr, acc[mi][ni]);
        }
    }

#pragma unroll
    for (int mi = 0; mi < 4; ++mi)
#pragma unroll
        for (int ni = 0; ni < 2; ++ni) {
            const int n = n0 + 16 * ni + col;
            const float bv = bias[n];
#pragma unroll
            for (int r = 0; r < 8; ++r) {
                const int m = m0 + 16 * mi + r + 8 * half;
                Out[(size_t)m * 1024 + n] = acc[mi][ni][r] + bv;
            }
        }
}

// =====================================================================
extern "C" void kernel_launch(void* const* d_in, const int* in_sizes, int n_in,
                              void* d_out, int out_size, void* d_ws, size_t ws_size,
                              hipStream_t stream) {
    const float* q  = (const float*)d_in[0];
    const float* k  = (const float*)d_in[1];
    const float* v  = (const float*)d_in[2];
    const float* Wq = (const float*)d_in[3];
    const float* bq = (const float*)d_in[4];
    const float* Wk = (const float*)d_in[5];
    const float* bk = (const float*)d_in[6];
    const float* Wv = (const float*)d_in[7];
    const float* bv = (const float*)d_in[8];
    const float* Wo = (const float*)d_in[9];
    const float* bo = (const float*)d_in[10];
    float* out = (float*)d_out;

    const size_t NEL = (size_t)4 * 4096 * 1024;   // 16,777,216 activation elements
    const size_t WEL = (size_t)1024 * 1024;       // weight elements
    unsigned short* Qb  = (unsigned short*)d_ws;
    unsigned short* Kb  = Qb + NEL;
    unsigned short* Vt  = Kb + NEL;
    unsigned short* At  = Vt + NEL;
    unsigned short* Xt  = At + NEL;               // staging for fp32->bf16 activations
    unsigned short* Wqb = Xt + NEL;
    unsigned short* Wkb = Wqb + WEL;
    unsigned short* Wvb = Wkb + WEL;
    unsigned short* Wob = Wvb + WEL;

    const int cvtW = (int)(WEL / (256 * 8));      // 512 blocks
    const int cvtX = (int)(NEL / (256 * 8));      // 8192 blocks

    // weights -> bf16 (once)
    cvt_bf16_kernel<<<cvtW, 256, 0, stream>>>(Wq, Wqb);
    cvt_bf16_kernel<<<cvtW, 256, 0, stream>>>(Wk, Wkb);
    cvt_bf16_kernel<<<cvtW, 256, 0, stream>>>(Wv, Wvb);
    cvt_bf16_kernel<<<cvtW, 256, 0, stream>>>(Wo, Wob);

    dim3 gProj(16384 / 128, 1024 / 128);          // 128 x 8 workgroups

    // Q projection (pre-scaled by 1/sqrt(64))
    cvt_bf16_kernel<<<cvtX, 256, 0, stream>>>(q, Xt);
    proj_qkv_kernel<<<gProj, 256, 0, stream>>>(Xt, Wqb, bq, Qb, 0.125f, 0);
    // K projection
    cvt_bf16_kernel<<<cvtX, 256, 0, stream>>>(k, Xt);
    proj_qkv_kernel<<<gProj, 256, 0, stream>>>(Xt, Wkb, bk, Kb, 1.0f, 0);
    // V projection (transposed output)
    cvt_bf16_kernel<<<cvtX, 256, 0, stream>>>(v, Xt);
    proj_qkv_kernel<<<gProj, 256, 0, stream>>>(Xt, Wvb, bv, Vt, 1.0f, 1);

    dim3 gAttn(32, 16, 4);                        // (block, head, batch)
    attn_kernel<<<gAttn, 128, 0, stream>>>(Qb, Kb, Vt, At);

    proj_out_kernel<<<gProj, 256, 0, stream>>>(At, Wob, bo, out);
}
